// SparseDeltaModule_11914239279727
// MI455X (gfx1250) — matrix operand order repaired and checked by reading, not compiled, not run on hardware
//
#include <hip/hip_runtime.h>
#include <hip/hip_bf16.h>
#include <cstdint>

// ---------------------------------------------------------------------------
// TopK-SAE forward for MI455X (gfx1250, wave32, WMMA).
//   B=4096 tokens, D=4096 input dim, L=32768 latents, K=32 topk.
// Pipeline:
//   1) convert inputs + W_enc to bf16 (workspace)
//   2) transpose+convert W_dec -> Wt[L][D] bf16 (workspace)
//   3) bf16 WMMA GEMM + bias + SiLU -> dense acts (staged in sparse_latents
//      output region); double-buffered async global->LDS copies; grid ordered
//      M-fast so each W_enc N-strip stays L2-resident
//   4) per-row iterative top-32 (LDS argmax tree)
//   5) decode: delta[b,:] = sum_j val_j * Wt[idx_j][:]
//   6) zero sparse region + scatter topk values
// ---------------------------------------------------------------------------

#define B_DIM 4096
#define D_DIM 4096
#define L_DIM 32768
#define TOPK  32

typedef __attribute__((ext_vector_type(16))) __bf16       v16bf;
typedef __attribute__((ext_vector_type(8)))  float        v8f;
typedef __attribute__((ext_vector_type(4)))  unsigned int u32x4;
typedef __attribute__((ext_vector_type(4)))  int          i32x4;
typedef __attribute__((ext_vector_type(4)))  float        f32x4;

union Frag { v16bf v; u32x4 q[2]; };

__device__ __forceinline__ unsigned int f2bf(float f) {
  unsigned int u = __float_as_uint(f);
  return (u + 0x7FFFu + ((u >> 16) & 1u)) >> 16;   // round-to-nearest-even
}
__device__ __forceinline__ float bf_lo(unsigned int p) {
  return __uint_as_float((p & 0xFFFFu) << 16);
}
__device__ __forceinline__ float bf_hi(unsigned int p) {
  return __uint_as_float(p & 0xFFFF0000u);
}

// ---- async global->LDS (CDNA5 GLOBAL_LOAD_ASYNC_TO_LDS_B128), guarded -----
#if __has_builtin(__builtin_amdgcn_global_load_async_to_lds_b128)
#define HAVE_ASYNC_LDS 1
typedef __attribute__((address_space(1))) i32x4 as1_i32x4;   // global int4
typedef __attribute__((address_space(3))) i32x4 as3_i32x4;   // LDS int4
__device__ __forceinline__ void async_cp16(void* lds, const void* g) {
  __builtin_amdgcn_global_load_async_to_lds_b128((as1_i32x4*)g,
                                                 (as3_i32x4*)lds, 0, 0);
}
__device__ __forceinline__ void async_wait0() {
#if __has_builtin(__builtin_amdgcn_s_wait_asynccnt)
  __builtin_amdgcn_s_wait_asynccnt(0);
#else
  asm volatile("s_wait_asynccnt 0x0" ::: "memory");
#endif
}
#else
#define HAVE_ASYNC_LDS 0
#endif

// ---------------------------------------------------------------- convert ---
__global__ __launch_bounds__(256) void cvt_f32_bf16(const float* __restrict__ src,
                                                    unsigned short* __restrict__ dst,
                                                    size_t n8) {
  size_t i = (size_t)blockIdx.x * 256 + threadIdx.x;
  if (i >= n8) return;
  const f32x4* s = (const f32x4*)src + i * 2;
  f32x4 a = s[0], b = s[1];
  u32x4 o;
  o[0] = f2bf(a[0]) | (f2bf(a[1]) << 16);
  o[1] = f2bf(a[2]) | (f2bf(a[3]) << 16);
  o[2] = f2bf(b[0]) | (f2bf(b[1]) << 16);
  o[3] = f2bf(b[2]) | (f2bf(b[3]) << 16);
  ((u32x4*)dst)[i] = o;
}

// ---------------------------------------------- W_dec [D][L] -> Wt [L][D] ---
__global__ __launch_bounds__(256) void transpose_wdec(const float* __restrict__ Wd,
                                                      unsigned short* __restrict__ Wt) {
  __shared__ float tile[32][33];
  const int lb = blockIdx.x * 32;            // latent base
  const int db = blockIdx.y * 32;            // dim base
  const int tx = threadIdx.x & 31;
  const int ty = threadIdx.x >> 5;           // 0..7
#pragma unroll
  for (int s = 0; s < 4; ++s) {
    int d = db + ty + s * 8;
    tile[ty + s * 8][tx] = Wd[(size_t)d * L_DIM + lb + tx];
  }
  __syncthreads();
#pragma unroll
  for (int s = 0; s < 4; ++s) {
    int l = lb + ty + s * 8;
    Wt[(size_t)l * D_DIM + db + tx] = (unsigned short)f2bf(tile[tx][ty + s * 8]);
  }
}

// --------------------------------------------------- encoder WMMA GEMM -----
// C[m][n] = silu( sum_k X[m][k]*W_enc[n][k] + b[n] ),  bf16 inputs, f32 acc.
// Block tile 128x256, BK=32, double-buffered LDS. 8 waves: 2 along M x 4
// along N; wave tile 64x64 = 4x4 WMMA subtiles (16 wmma : 16 ds_load_b128).
#define BM 128
#define BN 256
#define BK 32

__global__ __launch_bounds__(256) void enc_gemm(const unsigned short* __restrict__ Abf,
                                                const unsigned short* __restrict__ Bbf,
                                                const float* __restrict__ b_enc,
                                                float* __restrict__ acts) {
  __shared__ unsigned short lA[2][BM * BK];   // 2 x 8 KB
  __shared__ unsigned short lB[2][BN * BK];   // 2 x 16 KB

  const int t    = threadIdx.x;
  const int lane = t & 31;
  const int wave = t >> 5;
  const int wm   = wave & 1;        // 0..1 -> M offset wm*64
  const int wn   = wave >> 1;       // 0..3 -> N offset wn*64
  const int m0   = blockIdx.x * BM; // M fast-varying: W_enc strip stays in L2
  const int n0   = blockIdx.y * BN;

  const int hh = lane >> 4;         // lane half (0/1)
  const int lr = lane & 15;
  const int aK0 = hh * 8;           // A frag: K = aK0..aK0+7 , 16+aK0..+7
  const int bK0 = hh * 8;           // B frag: K = bK0..bK0+7 , 16+bK0..+7

  v8f zero = {};
  v8f acc[4][4];
#pragma unroll
  for (int i = 0; i < 4; ++i)
#pragma unroll
    for (int j = 0; j < 4; ++j) acc[i][j] = zero;

  // stage one BK-slice of A (512 x 16B chunks) and B (1024 x 16B chunks)
  auto issue = [&](int s, int k0) {
#pragma unroll
    for (int c = 0; c < 2; ++c) {             // A: 2 chunks / thread
      int chunk = t + c * 256;
      int row   = chunk >> 2;
      int ko    = (chunk & 3) * 8;
#if HAVE_ASYNC_LDS
      async_cp16(&lA[s][row * BK + ko],
                 &Abf[(size_t)(m0 + row) * D_DIM + k0 + ko]);
#else
      *(u32x4*)&lA[s][row * BK + ko] =
          *(const u32x4*)&Abf[(size_t)(m0 + row) * D_DIM + k0 + ko];
#endif
    }
#pragma unroll
    for (int c = 0; c < 4; ++c) {             // B: 4 chunks / thread
      int chunk = t + c * 256;
      int row   = chunk >> 2;
      int ko    = (chunk & 3) * 8;
#if HAVE_ASYNC_LDS
      async_cp16(&lB[s][row * BK + ko],
                 &Bbf[(size_t)(n0 + row) * D_DIM + k0 + ko]);
#else
      *(u32x4*)&lB[s][row * BK + ko] =
          *(const u32x4*)&Bbf[(size_t)(n0 + row) * D_DIM + k0 + ko];
#endif
    }
  };

  issue(0, 0);
#if HAVE_ASYNC_LDS
  async_wait0();
#endif
  __syncthreads();

  for (int k0 = 0; k0 < D_DIM; k0 += BK) {
    const int cur = (k0 >> 5) & 1;
    if (k0 + BK < D_DIM) issue(cur ^ 1, k0 + BK);   // overlap copy with WMMA

    Frag fa[4], fb[4];
#pragma unroll
    for (int i = 0; i < 4; ++i) {
      int am = wm * 64 + i * 16 + lr;
      fa[i].q[0] = *(const u32x4*)&lA[cur][am * BK + aK0];
      fa[i].q[1] = *(const u32x4*)&lA[cur][am * BK + 16 + aK0];
    }
#pragma unroll
    for (int j = 0; j < 4; ++j) {
      int bn = wn * 64 + j * 16 + lr;
      fb[j].q[0] = *(const u32x4*)&lB[cur][bn * BK + bK0];
      fb[j].q[1] = *(const u32x4*)&lB[cur][bn * BK + 16 + bK0];
    }
#pragma unroll
    for (int i = 0; i < 4; ++i)
#pragma unroll
      for (int j = 0; j < 4; ++j)
        acc[i][j] = __builtin_amdgcn_wmma_f32_16x16x32_bf16(
            false, fa[i].v, false, fb[j].v, (short)0, acc[i][j], false, false);

#if HAVE_ASYNC_LDS
    async_wait0();
#endif
    __syncthreads();
  }

  // epilogue: +bias, SiLU, store dense activations
#pragma unroll
  for (int i = 0; i < 4; ++i) {
    int gm_base = m0 + wm * 64 + i * 16 + hh * 8;
#pragma unroll
    for (int j = 0; j < 4; ++j) {
      int gn = n0 + wn * 64 + j * 16 + lr;
      float bias = b_enc[gn];
#pragma unroll
      for (int r = 0; r < 8; ++r) {
        float x = acc[i][j][r] + bias;
        float s = x * (1.0f / (1.0f + __expf(-x)));
        acts[(size_t)(gm_base + r) * L_DIM + gn] = s;
      }
    }
  }
}

// ----------------------------------------------------------------- top-k ---
__global__ __launch_bounds__(256) void topk_kernel(float* __restrict__ acts,
                                                   int* __restrict__ out_idx,
                                                   float* __restrict__ out_vals) {
  const int row = blockIdx.x;
  const int t   = threadIdx.x;
  float* a = acts + (size_t)row * L_DIM;

  __shared__ float lmax[256]; __shared__ int larg[256];
  __shared__ float rmax[256]; __shared__ int rarg[256]; __shared__ int rown[256];
  __shared__ int wi_s, wo_s;

  const float NEG = -__builtin_inff();
  float m = NEG; int mi = 0;
  for (int i = t; i < L_DIM; i += 256) {      // coalesced strided scan
    float v = a[i];
    if (v > m) { m = v; mi = i; }
  }
  lmax[t] = m; larg[t] = mi;
  __syncthreads();

  for (int iter = 0; iter < TOPK; ++iter) {
    rmax[t] = lmax[t]; rarg[t] = larg[t]; rown[t] = t;
    __syncthreads();
    for (int s = 128; s > 0; s >>= 1) {
      if (t < s && rmax[t + s] > rmax[t]) {
        rmax[t] = rmax[t + s]; rarg[t] = rarg[t + s]; rown[t] = rown[t + s];
      }
      __syncthreads();
    }
    if (t == 0) {
      wi_s = rarg[0]; wo_s = rown[0];
      out_vals[row * TOPK + iter] = rmax[0];
      out_idx[row * TOPK + iter]  = rarg[0];
    }
    __syncthreads();
    if (t == wo_s) {                           // only owner rescans its chunk
      a[wi_s] = NEG;
      float nm = NEG; int nmi = 0;
      for (int i = t; i < L_DIM; i += 256) {
        float v = a[i];
        if (v > nm) { nm = v; nmi = i; }
      }
      lmax[t] = nm; larg[t] = nmi;
    }
    __syncthreads();
  }
}

// ---------------------------------------------------------------- decode ---
__global__ __launch_bounds__(256) void decode_kernel(const unsigned short* __restrict__ Wt,
                                                     const float* __restrict__ vals,
                                                     const int* __restrict__ idx,
                                                     float* __restrict__ delta) {
  const int row = blockIdx.x;
  const int t   = threadIdx.x;
  __shared__ float sv[TOPK]; __shared__ int si[TOPK];
  if (t < TOPK) { sv[t] = vals[row * TOPK + t]; si[t] = idx[row * TOPK + t]; }
  __syncthreads();

  float accd[16];
#pragma unroll
  for (int e = 0; e < 16; ++e) accd[e] = 0.0f;
  const int d0 = t * 16;                       // contiguous 16 outputs / thread

  for (int j = 0; j < TOPK; ++j) {
    float v = sv[j];
    const unsigned short* wrow = Wt + (size_t)si[j] * D_DIM + d0;
    u32x4 q0 = *(const u32x4*)wrow;
    u32x4 q1 = *(const u32x4*)(wrow + 8);
#pragma unroll
    for (int w = 0; w < 4; ++w) {
      accd[2 * w + 0] += v * bf_lo(q0[w]);
      accd[2 * w + 1] += v * bf_hi(q0[w]);
      accd[2 * w + 8] += v * bf_lo(q1[w]);
      accd[2 * w + 9] += v * bf_hi(q1[w]);
    }
  }

  f32x4* dst = (f32x4*)&delta[(size_t)row * D_DIM + d0];
#pragma unroll
  for (int w = 0; w < 4; ++w) {
    f32x4 o; o[0] = accd[4 * w]; o[1] = accd[4 * w + 1];
    o[2] = accd[4 * w + 2]; o[3] = accd[4 * w + 3];
    dst[w] = o;
  }
}

// ------------------------------------------------------- zero + scatter ----
__global__ __launch_bounds__(256) void zero_kernel(f32x4* __restrict__ p, size_t n4) {
  size_t i = (size_t)blockIdx.x * 256 + threadIdx.x;
  f32x4 z = {};
  if (i < n4) p[i] = z;
}

__global__ void scatter_kernel(const float* __restrict__ vals,
                               const int* __restrict__ idx,
                               float* __restrict__ sparse) {
  int row = blockIdx.x;
  int t   = threadIdx.x;   // 0..31
  sparse[(size_t)row * L_DIM + idx[row * TOPK + t]] = vals[row * TOPK + t];
}

// ---------------------------------------------------------------------------
extern "C" void kernel_launch(void* const* d_in, const int* in_sizes, int n_in,
                              void* d_out, int out_size, void* d_ws, size_t ws_size,
                              hipStream_t stream) {
  (void)in_sizes; (void)n_in; (void)out_size; (void)ws_size;
  const float* inputs = (const float*)d_in[0];   // [B, D]
  const float* W_enc  = (const float*)d_in[1];   // [L, D]
  const float* b_enc  = (const float*)d_in[2];   // [L]
  const float* W_dec  = (const float*)d_in[3];   // [D, L]

  float* out_delta  = (float*)d_out;                                  // [B, D]
  float* out_sparse = out_delta + (size_t)B_DIM * D_DIM;              // [B, L]
  int*   out_idx    = (int*)(out_sparse + (size_t)B_DIM * L_DIM);     // [B, K]
  float* out_vals   = (float*)(out_idx + (size_t)B_DIM * TOPK);       // [B, K]

  unsigned short* ws_x  = (unsigned short*)d_ws;                      // X bf16   32MB
  unsigned short* ws_we = ws_x  + (size_t)B_DIM * D_DIM;              // Wenc bf16 256MB
  unsigned short* ws_wt = ws_we + (size_t)L_DIM * D_DIM;              // WdecT bf16 256MB

  // 1) precision conversion
  {
    size_t n8 = (size_t)B_DIM * D_DIM / 8;
    cvt_f32_bf16<<<dim3((unsigned)((n8 + 255) / 256)), dim3(256), 0, stream>>>(inputs, ws_x, n8);
  }
  {
    size_t n8 = (size_t)L_DIM * D_DIM / 8;
    cvt_f32_bf16<<<dim3((unsigned)((n8 + 255) / 256)), dim3(256), 0, stream>>>(W_enc, ws_we, n8);
  }
  // 2) decoder transpose
  transpose_wdec<<<dim3(L_DIM / 32, D_DIM / 32), dim3(256), 0, stream>>>(W_dec, ws_wt);

  // 3) encoder GEMM + SiLU (dense acts staged in sparse output region)
  enc_gemm<<<dim3(B_DIM / BM, L_DIM / BN), dim3(256), 0, stream>>>(ws_x, ws_we, b_enc, out_sparse);

  // 4) top-32 per row
  topk_kernel<<<dim3(B_DIM), dim3(256), 0, stream>>>(out_sparse, out_idx, out_vals);

  // 5) decode
  decode_kernel<<<dim3(B_DIM), dim3(256), 0, stream>>>(ws_wt, out_vals, out_idx, out_delta);

  // 6) sparsify output region
  {
    size_t n4 = (size_t)B_DIM * L_DIM / 4;
    zero_kernel<<<dim3((unsigned)((n4 + 255) / 256)), dim3(256), 0, stream>>>((f32x4*)out_sparse, n4);
  }
  scatter_kernel<<<dim3(B_DIM), dim3(TOPK), 0, stream>>>(out_vals, out_idx, out_sparse);
}
